// CrossModalAttention_44057774522559
// MI455X (gfx1250) — matrix operand order repaired and checked by reading, not compiled, not run on hardware
//
#include <hip/hip_runtime.h>
#include <hip/hip_bf16.h>

typedef __attribute__((ext_vector_type(16))) __bf16 v16bf;
typedef __attribute__((ext_vector_type(8)))  __bf16 v8bf;
typedef __attribute__((ext_vector_type(8)))  float  v8f;
typedef __attribute__((ext_vector_type(4)))  int    v4i;

#define K_  4
#define B_  1024
#define C_  128
#define T_  64
#define D_  64
#define H_  4
#define HD_ 16
#define EPS_ 1e-5f

#define LDC 72   // bf16 LDS row pitch (144B: keeps 16B alignment, offsets banks)
#define LDO 68   // f32 LDS row pitch

#define AS_GLOBAL __attribute__((address_space(1)))
#define AS_LDS    __attribute__((address_space(3)))

// ---------------------------------------------------------------------------
// Fragment helpers (wave32, v_wmma_f32_16x16x32_bf16).
// A-fragment (16x32 bf16): lane holds two contiguous 8-element K runs:
//   elems 0..7  -> K = k0 + khalf + 0..7
//   elems 8..15 -> K = k0 + khalf + 8 + 0..7       (khalf = (lane>=16)*16)
// so each run is one 16-byte LDS load.
__device__ __forceinline__ v16bf load_a_rm(const __bf16* A, int ld, int row0, int k0) {
  const int lane  = threadIdx.x & 31;
  const int khalf = (lane >> 4) * 16;
  const __bf16* p = A + (row0 + (lane & 15)) * ld + k0 + khalf;
  v8bf lo = *(const v8bf*)(p);
  v8bf hi = *(const v8bf*)(p + 8);
  return __builtin_shufflevector(lo, hi, 0, 1, 2, 3, 4, 5, 6, 7,
                                 8, 9, 10, 11, 12, 13, 14, 15);
}

// A-fragment with strided row map (row = (row0+lane%16)*mul + add): used by the
// q-GEMM to gather ctx rows 4*m + i.
__device__ __forceinline__ v16bf load_a_strided(const __bf16* A, int ld, int row0,
                                                int mul, int add, int k0) {
  const int lane  = threadIdx.x & 31;
  const int khalf = (lane >> 4) * 16;
  const __bf16* p = A + ((row0 + (lane & 15)) * mul + add) * ld + k0 + khalf;
  v8bf lo = *(const v8bf*)(p);
  v8bf hi = *(const v8bf*)(p + 8);
  return __builtin_shufflevector(lo, hi, 0, 1, 2, 3, 4, 5, 6, 7,
                                 8, 9, 10, 11, 12, 13, 14, 15);
}

// Packed B-fragment: per-lane 16 contiguous bf16 (32B) -> single vector load.
__device__ __forceinline__ v16bf load_frag32(const __bf16* F) {
  return *(const v16bf*)(F + (threadIdx.x & 31) * 16);
}

// ---------------------------------------------------------------------------
// Setup: pack all q/k/v/o weight B-fragments (B[k][n] = W[n][k]) into
// per-lane-contiguous order: Wf[((((i*4+mat)*4+dT)*2+kkc)*32+lane)*16+e]
//   n = dT*16 + lane%16,  k = kkc*32 + (lane/16)*16 + e
__global__ __launch_bounds__(256) void pack_wfrag_kernel(
    const float* __restrict__ wq, const float* __restrict__ wk,
    const float* __restrict__ wv, const float* __restrict__ wo,
    __bf16* __restrict__ Wf) {
  const float* Ws[4] = {wq, wk, wv, wo};
  const int total = K_ * 4 * 4 * 2 * 32 * 16;
  for (int idx = blockIdx.x * 256 + threadIdx.x; idx < total; idx += gridDim.x * 256) {
    const int e    = idx & 15;
    const int lane = (idx >> 4) & 31;
    const int kkc  = (idx >> 9) & 1;
    const int dT   = (idx >> 10) & 3;
    const int mat  = (idx >> 12) & 3;
    const int i    = idx >> 14;
    const int n = dT * 16 + (lane & 15);
    const int k = kkc * 32 + (lane >> 4) * 16 + e;
    Wf[idx] = (__bf16)Ws[mat][((size_t)i * D_ + n) * D_ + k];
  }
}

__device__ __forceinline__ const __bf16* wfrag(const __bf16* Wf, int i, int mat,
                                               int dT16, int kkc) {
  return Wf + (((((i * 4 + mat) * 4 + dT16) * 2 + kkc)) << 9);
}

// ---------------------------------------------------------------------------
// Stage 1: proj[k,b,d,t] -> ctx workspace P[b][k*64+d][t] (bf16).
// feats staged into LDS directly in B-fragment order so B loads are b128s.
__global__ __launch_bounds__(256) void proj_kernel(const float* __restrict__ feats,
                                                   const float* __restrict__ proj_w,
                                                   const float* __restrict__ proj_b,
                                                   __bf16* __restrict__ P) {
  __shared__ __align__(32) __bf16 ldsW[D_ * C_];   // A: [d][c] row-major
  __shared__ __align__(32) __bf16 ldsF[C_ * T_];   // B fragments: [kkc][tT][lane][e]
  const int tid = threadIdx.x;
  const int k   = blockIdx.x >> 10;
  const int b   = blockIdx.x & (B_ - 1);

  const float* wsrc = proj_w + (size_t)k * D_ * C_;
  for (int i = tid; i < D_ * C_; i += 256) ldsW[i] = (__bf16)wsrc[i];
  const float* fsrc = feats + (size_t)(k * B_ + b) * C_ * T_;
  for (int i = tid; i < C_ * T_; i += 256) {
    const int c = i >> 6, t = i & 63;
    const int kkc = c >> 5, cc = c & 31;
    const int lane = (t & 15) + ((cc >> 4) << 4);
    const int e = cc & 15;
    ldsF[(((kkc * 4 + (t >> 4)) * 32 + lane) << 4) + e] = (__bf16)fsrc[i];
  }
  __syncthreads();

  const int wave = tid >> 5;
  const int lane = tid & 31;
  const int dT  = (wave >> 1) * 16;   // A rows fixed per wave
  const int tT0 = (wave & 1) * 32;    // two output col-tiles per wave

  v16bf a[4];
#pragma unroll
  for (int kc = 0; kc < 4; ++kc) a[kc] = load_a_rm(ldsW, C_, dT, kc * 32);

#pragma unroll
  for (int s = 0; s < 2; ++s) {
    const int tT = tT0 + s * 16;
    v8f acc = {};
#pragma unroll
    for (int kc = 0; kc < 4; ++kc) {
      v16bf bb = load_frag32(ldsF + (((kc * 4 + (tT >> 4)) * 32) << 4));
      acc = __builtin_amdgcn_wmma_f32_16x16x32_bf16(false, a[kc], false, bb, (short)0,
                                                    acc, false, false);
    }
    const int n  = tT + (lane & 15);
    const int rb = dT + (lane >> 4) * 8;
#pragma unroll
    for (int r = 0; r < 8; ++r) {
      const int d = rb + r;
      P[((size_t)b * (K_ * D_) + (k * D_ + d)) * T_ + n] =
          (__bf16)(acc[r] + proj_b[k * D_ + d]);
    }
  }
}

// ---------------------------------------------------------------------------
// Stage 2: fused q/k/v GEMMs + attention + out-proj + LN per b.
__global__ __launch_bounds__(256) void attn_kernel(const __bf16* __restrict__ P,
    const __bf16* __restrict__ Wf,
    const float* __restrict__ bq, const float* __restrict__ bk,
    const float* __restrict__ bv, const float* __restrict__ bo,
    const float* __restrict__ ng, const float* __restrict__ nb,
    const float* __restrict__ tfw, float* __restrict__ out) {
  __shared__ __align__(32) __bf16 ctx[(K_ * D_) * LDC];  // 256 x 64 ctx rows
  __shared__ __align__(32) __bf16 qb[D_ * LDC];          // q, then attention out o
  __shared__ __align__(32) __bf16 kb[(K_ * D_) * LDC];
  __shared__ __align__(32) __bf16 vb[(K_ * D_) * LDC];
  __shared__ __align__(16) float  obuf[D_ * LDO];

  const int tid  = threadIdx.x;
  const int b    = blockIdx.x;
  const int wave = tid >> 5;
  const int lane = tid & 31;

  // ---- stage ctx (256 rows x 64 bf16 = 32KB) into LDS ----
#if __has_builtin(__builtin_amdgcn_global_load_async_to_lds_b128)
  // CDNA5 async DMA path: per-lane 16B global->LDS transfers, ASYNCcnt-tracked.
  {
    AS_GLOBAL char* src =
        (AS_GLOBAL char*)(P + (size_t)b * (K_ * D_) * T_);
    AS_LDS char* dst = (AS_LDS char*)ctx;
    for (int g = tid; g < (K_ * D_) * T_ * 2 / 16; g += 256) {
      const int row = g >> 3;            // 8 x 16B chunks per 128B row
      const int ch  = g & 7;
      __builtin_amdgcn_global_load_async_to_lds_b128(
          (AS_GLOBAL v4i*)(src + (size_t)g * 16),
          (AS_LDS v4i*)(dst + row * (LDC * 2) + ch * 16), 0, 0);
    }
#if __has_builtin(__builtin_amdgcn_s_wait_asynccnt)
    __builtin_amdgcn_s_wait_asynccnt(0);
#else
    asm volatile("s_wait_asynccnt 0" ::: "memory");
#endif
  }
#else
  // fallback: staged through VGPRs, 2 bf16 per u32, coalesced
  {
    const unsigned* Pg = (const unsigned*)(P + (size_t)b * (K_ * D_) * T_);
    unsigned* ctxU = (unsigned*)ctx;
    for (int i = tid; i < (K_ * D_) * T_ / 2; i += 256)
      ctxU[((i >> 5) * LDC) / 2 + (i & 31)] = Pg[i];
  }
#endif
  __syncthreads();

  float wsm[K_];
  {
    float mx = tfw[0];
    for (int i = 1; i < K_; ++i) mx = fmaxf(mx, tfw[i]);
    float den = 0.f;
    for (int i = 0; i < K_; ++i) { wsm[i] = __expf(tfw[i] - mx); den += wsm[i]; }
    for (int i = 0; i < K_; ++i) wsm[i] /= den;
  }

  for (int i = 0; i < K_; ++i) {
    const float* bq_g = bq + i * D_;
    const float* bk_g = bk + i * D_;
    const float* bv_g = bv + i * D_;
    const float* bo_g = bo + i * D_;
    const float* ng_g = ng + i * D_;
    const float* nb_g = nb + i * D_;

    // ---- q = ctx[rows 4m+i] @ wq^T + bq (64x64): A once, 2 col-tiles ----
    {
      const int mT  = (wave >> 1) * 16;
      const int dT0 = (wave & 1) * 32;
      v16bf aq[2];
#pragma unroll
      for (int kc = 0; kc < 2; ++kc)
        aq[kc] = load_a_strided(ctx, LDC, mT, 4, i, kc * 32);
#pragma unroll
      for (int sd = 0; sd < 2; ++sd) {
        const int dT = dT0 + sd * 16;
        v8f acc = {};
#pragma unroll
        for (int kc = 0; kc < 2; ++kc) {
          v16bf bb = load_frag32(wfrag(Wf, i, 0, dT >> 4, kc));
          acc = __builtin_amdgcn_wmma_f32_16x16x32_bf16(false, aq[kc], false, bb,
                                                        (short)0, acc, false, false);
        }
        const int n  = dT + (lane & 15);
        const int rb = mT + (lane >> 4) * 8;
        const float bias = bq_g[n];
#pragma unroll
        for (int r = 0; r < 8; ++r) qb[(rb + r) * LDC + n] = (__bf16)(acc[r] + bias);
      }
    }

    // ---- k,v = ctx[256 rows] @ w^T + bias: A frags shared, B per dT ----
    {
      v16bf aj[2][2];
#pragma unroll
      for (int jt = 0; jt < 2; ++jt)
#pragma unroll
        for (int kc = 0; kc < 2; ++kc)
          aj[jt][kc] = load_a_rm(ctx, LDC, (wave * 2 + jt) * 16, kc * 32);
#pragma unroll
      for (int dT4 = 0; dT4 < 4; ++dT4) {
        v16bf bk0 = load_frag32(wfrag(Wf, i, 1, dT4, 0));
        v16bf bk1 = load_frag32(wfrag(Wf, i, 1, dT4, 1));
        v16bf bv0 = load_frag32(wfrag(Wf, i, 2, dT4, 0));
        v16bf bv1 = load_frag32(wfrag(Wf, i, 2, dT4, 1));
        const int n = dT4 * 16 + (lane & 15);
        const float biask = bk_g[n], biasv = bv_g[n];
#pragma unroll
        for (int jt = 0; jt < 2; ++jt) {
          v8f ak = {}, av = {};
          ak = __builtin_amdgcn_wmma_f32_16x16x32_bf16(false, aj[jt][0], false, bk0,
                                                       (short)0, ak, false, false);
          ak = __builtin_amdgcn_wmma_f32_16x16x32_bf16(false, aj[jt][1], false, bk1,
                                                       (short)0, ak, false, false);
          av = __builtin_amdgcn_wmma_f32_16x16x32_bf16(false, aj[jt][0], false, bv0,
                                                       (short)0, av, false, false);
          av = __builtin_amdgcn_wmma_f32_16x16x32_bf16(false, aj[jt][1], false, bv1,
                                                       (short)0, av, false, false);
          const int rb = (wave * 2 + jt) * 16 + (lane >> 4) * 8;
#pragma unroll
          for (int r = 0; r < 8; ++r) {
            kb[(rb + r) * LDC + n] = (__bf16)(ak[r] + biask);
            vb[(rb + r) * LDC + n] = (__bf16)(av[r] + biasv);
          }
        }
      }
    }
    __syncthreads();

    // ---- tiny attention: one (m,h) per thread, vectorized LDS reads ----
    {
      const int m = tid >> 2;
      const int h = tid & 3;
      const __bf16* qp = qb + m * LDC + h * HD_;
      float qv[HD_];
      {
        v8bf q0 = *(const v8bf*)qp, q1 = *(const v8bf*)(qp + 8);
#pragma unroll
        for (int e = 0; e < 8; ++e) { qv[e] = (float)q0[e]; qv[8 + e] = (float)q1[e]; }
      }
      float sc[K_];
      float mx = -1e30f;
#pragma unroll
      for (int c = 0; c < K_; ++c) {
        const __bf16* kp = kb + (4 * m + c) * LDC + h * HD_;
        v8bf k0 = *(const v8bf*)kp, k1 = *(const v8bf*)(kp + 8);
        float s = 0.f;
#pragma unroll
        for (int e = 0; e < 8; ++e) s += qv[e] * (float)k0[e] + qv[8 + e] * (float)k1[e];
        sc[c] = s * 0.25f;                       // 1/sqrt(HD)
        mx = fmaxf(mx, sc[c]);
      }
      float den = 0.f;
#pragma unroll
      for (int c = 0; c < K_; ++c) { sc[c] = __expf(sc[c] - mx); den += sc[c]; }
      const float inv = 1.f / den;
      float ov[HD_];
#pragma unroll
      for (int e = 0; e < HD_; ++e) ov[e] = 0.f;
#pragma unroll
      for (int c = 0; c < K_; ++c) {
        const __bf16* vp = vb + (4 * m + c) * LDC + h * HD_;
        v8bf v0 = *(const v8bf*)vp, v1 = *(const v8bf*)(vp + 8);
        const float a = sc[c];
#pragma unroll
        for (int e = 0; e < 8; ++e) { ov[e] += a * (float)v0[e]; ov[8 + e] += a * (float)v1[e]; }
      }
      v8bf o0, o1;
#pragma unroll
      for (int e = 0; e < 8; ++e) { o0[e] = (__bf16)(ov[e] * inv); o1[e] = (__bf16)(ov[8 + e] * inv); }
      *(v8bf*)(qb + m * LDC + h * HD_) = o0;
      *(v8bf*)(qb + m * LDC + h * HD_ + 8) = o1;
    }
    __syncthreads();

    // ---- o @ wo^T + bo + residual(q_in) -> obuf ----
    {
      const int mT  = (wave >> 1) * 16;
      const int dT0 = (wave & 1) * 32;
      v16bf ao[2];
#pragma unroll
      for (int kc = 0; kc < 2; ++kc) ao[kc] = load_a_rm(qb, LDC, mT, kc * 32);
#pragma unroll
      for (int sd = 0; sd < 2; ++sd) {
        const int dT = dT0 + sd * 16;
        v8f acc = {};
#pragma unroll
        for (int kc = 0; kc < 2; ++kc) {
          v16bf bb = load_frag32(wfrag(Wf, i, 3, dT >> 4, kc));
          acc = __builtin_amdgcn_wmma_f32_16x16x32_bf16(false, ao[kc], false, bb,
                                                        (short)0, acc, false, false);
        }
        const int n  = dT + (lane & 15);
        const int rb = mT + (lane >> 4) * 8;
        const float bias = bo_g[n];
#pragma unroll
        for (int r = 0; r < 8; ++r) {
          const int m = rb + r;
          obuf[m * LDO + n] = acc[r] + bias + (float)ctx[(4 * m + i) * LDC + n];
        }
      }
    }
    __syncthreads();

    // ---- per-row LN(64) * softmax(tfw)[i] -> d_out (pre final LN) ----
    {
      const int m = tid >> 2;
      const int part = tid & 3;
      float s = 0.f, s2 = 0.f;
#pragma unroll
      for (int e = 0; e < 16; ++e) {
        const float x = obuf[m * LDO + part * 16 + e];
        s += x; s2 += x * x;
      }
      s  += __shfl_xor(s, 1);  s  += __shfl_xor(s, 2);
      s2 += __shfl_xor(s2, 1); s2 += __shfl_xor(s2, 2);
      const float mean = s * (1.f / 64.f);
      const float var  = s2 * (1.f / 64.f) - mean * mean;
      const float rstd = rsqrtf(var + EPS_);
      const float wi = wsm[i];
#pragma unroll
      for (int e = 0; e < 16; ++e) {
        const int d = part * 16 + e;
        const float x = obuf[m * LDO + d];
        out[((size_t)b * D_ + m) * (K_ * D_) + i * D_ + d] =
            ((x - mean) * rstd * ng_g[d] + nb_g[d]) * wi;
      }
    }
    __syncthreads();
  }
}

// ---------------------------------------------------------------------------
// Stage 3: in-place final LN over 256-wide rows of d_out (one wave per row).
__global__ __launch_bounds__(256) void final_ln_kernel(float* __restrict__ out,
                                                       const float* __restrict__ fg,
                                                       const float* __restrict__ fb) {
  const int wave = threadIdx.x >> 5;
  const int lane = threadIdx.x & 31;
  float* p = out + ((size_t)blockIdx.x * 8 + wave) * (K_ * D_);
  float x[8];
  float s = 0.f, s2 = 0.f;
#pragma unroll
  for (int e = 0; e < 8; ++e) {
    x[e] = p[e * 32 + lane];
    s += x[e]; s2 += x[e] * x[e];
  }
#pragma unroll
  for (int off = 1; off < 32; off <<= 1) {
    s  += __shfl_xor(s, off);
    s2 += __shfl_xor(s2, off);
  }
  const float mean = s * (1.f / 256.f);
  const float var  = s2 * (1.f / 256.f) - mean * mean;
  const float rstd = rsqrtf(var + EPS_);
#pragma unroll
  for (int e = 0; e < 8; ++e) {
    const int j = e * 32 + lane;
    p[j] = (x[e] - mean) * rstd * fg[j] + fb[j];
  }
}

extern "C" void kernel_launch(void* const* d_in, const int* in_sizes, int n_in,
                              void* d_out, int out_size, void* d_ws, size_t ws_size,
                              hipStream_t stream) {
  (void)in_sizes; (void)n_in; (void)out_size; (void)ws_size;
  const float* feats  = (const float*)d_in[0];
  const float* proj_w = (const float*)d_in[1];
  const float* proj_b = (const float*)d_in[2];
  const float* wq  = (const float*)d_in[3];
  const float* wk  = (const float*)d_in[4];
  const float* wv  = (const float*)d_in[5];
  const float* bq  = (const float*)d_in[6];
  const float* bk  = (const float*)d_in[7];
  const float* bv  = (const float*)d_in[8];
  const float* wo  = (const float*)d_in[9];
  const float* bo  = (const float*)d_in[10];
  const float* ng  = (const float*)d_in[11];
  const float* nb  = (const float*)d_in[12];
  const float* tfw = (const float*)d_in[13];
  const float* fg  = (const float*)d_in[14];
  const float* fb  = (const float*)d_in[15];
  float* out = (float*)d_out;

  __bf16* P  = (__bf16*)d_ws;                               // 32 MiB ctx workspace
  __bf16* Wf = (__bf16*)((char*)d_ws + (size_t)B_ * (K_ * D_) * T_ * sizeof(__bf16));

  pack_wfrag_kernel<<<64, 256, 0, stream>>>(wq, wk, wv, wo, Wf);
  proj_kernel<<<K_ * B_, 256, 0, stream>>>(feats, proj_w, proj_b, P);
  attn_kernel<<<B_, 256, 0, stream>>>(P, Wf, bq, bk, bv, bo, ng, nb, tfw, out);
  final_ln_kernel<<<(B_ * D_) / 8, 256, 0, stream>>>(out, fg, fb);
}
